// Qwen3NextSparseMoeBlock_8495445311712
// MI455X (gfx1250) — compile-verified, matrix operand order checked by reading, not run on hardware
//
#include <hip/hip_runtime.h>
#include <hip/hip_bf16.h>
#include <math.h>

#define HIDDEN   2048
#define INTER    768
#define NEXP     16
#define NTOK     8192          // B*S = 4*2048
#define TILE_M   16

typedef __bf16 v16bf __attribute__((ext_vector_type(16)));
typedef __bf16 v4bf  __attribute__((ext_vector_type(4)));
typedef float  v8f   __attribute__((ext_vector_type(8)));

// ---------------------------------------------------------------------------
__device__ __forceinline__ v16bf cvt16(const float* __restrict__ p) {
    v16bf r;
#pragma unroll
    for (int j = 0; j < 16; ++j) r[j] = (__bf16)p[j];
    return r;
}

__device__ __forceinline__ v8f wmma_bf16(v16bf a, v16bf b, v8f c) {
    return __builtin_amdgcn_wmma_f32_16x16x32_bf16(
        false, a, false, b, (short)0, c, false, false);
}

// ---------------------------------------------------------------------------
// Kernel 0: zero output + expert counters
// ---------------------------------------------------------------------------
__global__ void zero_kernel(float* __restrict__ out, size_t n, int* __restrict__ cnt) {
    size_t i      = (size_t)blockIdx.x * blockDim.x + threadIdx.x;
    size_t stride = (size_t)gridDim.x * blockDim.x;
    for (; i < n; i += stride) out[i] = 0.0f;
    if (blockIdx.x == 0 && threadIdx.x < NEXP) cnt[threadIdx.x] = 0;
}

// ---------------------------------------------------------------------------
// Pre-conversion: x fp32 -> bf16 row-major
// ---------------------------------------------------------------------------
__global__ void convert_x_kernel(const float* __restrict__ src, __bf16* __restrict__ dst,
                                 size_t n4) {
    size_t i      = (size_t)blockIdx.x * blockDim.x + threadIdx.x;
    size_t stride = (size_t)gridDim.x * blockDim.x;
    for (; i < n4; i += stride) {
        const float4 v = reinterpret_cast<const float4*>(src)[i];
        v4bf o;
        o[0] = (__bf16)v.x; o[1] = (__bf16)v.y; o[2] = (__bf16)v.z; o[3] = (__bf16)v.w;
        *reinterpret_cast<v4bf*>(dst + i * 4) = o;
    }
}

// ---------------------------------------------------------------------------
// Pre-conversion + swizzle: weight fp32 [E][OUT][K] -> bf16 in WMMA fragment
// order: group index t = ((e*ntiles + ntile)*ksteps + kstep)*32 + lane,
// 16 K-contiguous elements per group. A wave's B-fragment load then becomes
// one lane-contiguous 1KB burst.
// ---------------------------------------------------------------------------
__global__ void swizzle_w_kernel(const float* __restrict__ src, __bf16* __restrict__ dst,
                                 int OUT, int K, size_t ngroups) {
    const int ksteps = K >> 5;
    const int ntiles = OUT >> 4;
    size_t t      = (size_t)blockIdx.x * blockDim.x + threadIdx.x;
    size_t stride = (size_t)gridDim.x * blockDim.x;
    for (; t < ngroups; t += stride) {
        const int lane = (int)(t & 31);
        size_t rest = t >> 5;
        const int ks    = (int)(rest % ksteps); rest /= ksteps;
        const int ntile = (int)(rest % ntiles);
        const size_t e  = rest / ntiles;
        const int row = ntile * 16 + (lane & 15);
        const int kb  = ks * 32 + ((lane >> 4) << 4);
        const float* s = src + (size_t)e * OUT * K + (size_t)row * K + kb;
        *reinterpret_cast<v16bf*>(dst + t * 16) = cvt16(s);
    }
}

// ---------------------------------------------------------------------------
// Kernel 1: routing (one wave32 per token)
// ---------------------------------------------------------------------------
__global__ __launch_bounds__(256) void route_kernel(
    const float* __restrict__ x, const float* __restrict__ gw,
    int* __restrict__ cnt, int* __restrict__ btok, float* __restrict__ bw) {

    const int wave  = threadIdx.x >> 5;
    const int lane  = threadIdx.x & 31;
    const int token = blockIdx.x * (blockDim.x >> 5) + wave;
    if (token >= NTOK) return;

    const float* xr = x + (size_t)token * HIDDEN;
    float acc[NEXP];
#pragma unroll
    for (int e = 0; e < NEXP; ++e) acc[e] = 0.0f;
    for (int h = lane; h < HIDDEN; h += 32) {
        const float xv = xr[h];
#pragma unroll
        for (int e = 0; e < NEXP; ++e) acc[e] += xv * gw[e * HIDDEN + h];
    }
#pragma unroll
    for (int off = 16; off >= 1; off >>= 1) {
#pragma unroll
        for (int e = 0; e < NEXP; ++e) acc[e] += __shfl_xor(acc[e], off, 32);
    }
    if (lane == 0) {
        int   e1 = 0, e2 = -1;
        float l1 = acc[0], l2 = -3.0e38f;
#pragma unroll
        for (int e = 1; e < NEXP; ++e) {
            if (acc[e] > l1)      { l2 = l1; e2 = e1; l1 = acc[e]; e1 = e; }
            else if (acc[e] > l2) { l2 = acc[e]; e2 = e; }
        }
        const float w1 = 1.0f / (1.0f + __expf(l2 - l1));
        const float w2 = 1.0f - w1;
        int p1 = atomicAdd(&cnt[e1], 1);
        btok[e1 * NTOK + p1] = token;  bw[e1 * NTOK + p1] = w1;
        int p2 = atomicAdd(&cnt[e2], 1);
        btok[e2 * NTOK + p2] = token;  bw[e2 * NTOK + p2] = w2;
    }
}

// ---------------------------------------------------------------------------
// Kernel 2: fused expert FFN for one (expert, 16-token tile), 8 waves.
//   PRE=true : bf16 x + fragment-swizzled bf16 weights (coalesced b128 loads)
//   PRE=false: fp32 sources, convert on the fly (fallback if ws too small)
// ---------------------------------------------------------------------------
template <bool PRE>
__global__ __launch_bounds__(256) void expert_ffn_kernel(
    const void* __restrict__ xsrc,
    const void* __restrict__ w_gate,
    const void* __restrict__ w_up,
    const void* __restrict__ w_down,
    const int*   __restrict__ cnt,
    const int*   __restrict__ btok,
    const float* __restrict__ bw,
    float*       __restrict__ out) {

    const int e    = blockIdx.y;
    const int tile = blockIdx.x;
    const int nt   = cnt[e];
    if (tile * TILE_M >= nt) return;

    __shared__ __align__(32) __bf16 xs[TILE_M][1024];
    __shared__ __align__(32) __bf16 act[TILE_M][INTER];
    __shared__ int   toks[TILE_M];
    __shared__ float twt[TILE_M];

    const int tid   = threadIdx.x;
    const int wave  = tid >> 5;
    const int lane  = tid & 31;
    const int ln16  = lane & 15;
    const int khalf = (lane >> 4) << 4;
    const int mhalf = (lane >> 4) << 3;

    if (tid < TILE_M) {
        const int idx = tile * TILE_M + tid;
        if (idx < nt) { toks[tid] = btok[e * NTOK + idx]; twt[tid] = bw[e * NTOK + idx]; }
        else          { toks[tid] = btok[e * NTOK + tile * TILE_M]; twt[tid] = 0.0f; }
    }
    __syncthreads();

    constexpr int KSGU = HIDDEN >> 5;   // 64 k-steps for gate/up
    constexpr int KSD  = INTER  >> 5;   // 24 k-steps for down

    v8f cg[6], cu[6];
#pragma unroll
    for (int i = 0; i < 6; ++i) { cg[i] = (v8f)0.0f; cu[i] = (v8f)0.0f; }

    // ---- gate & up projections, K split into two 1024 chunks -------------
    for (int kb = 0; kb < 2; ++kb) {
        if constexpr (PRE) {
            const __bf16* xb = (const __bf16*)xsrc;
#pragma unroll
            for (int it = 0; it < 8; ++it) {                 // 2048 x 16B chunks
                const int ui  = tid + it * 256;
                const int row = ui >> 7;                     // 128 chunks per row
                const int c8  = ui & 127;
                const uint4 v = *reinterpret_cast<const uint4*>(
                    xb + (size_t)toks[row] * HIDDEN + kb * 1024 + c8 * 8);
                *reinterpret_cast<uint4*>(&xs[row][c8 * 8]) = v;
            }
        } else {
            const float* xf = (const float*)xsrc;
#pragma unroll
            for (int it = 0; it < 16; ++it) {                // 4096 float4
                const int fi   = tid + it * 256;
                const int row  = fi >> 8;
                const int col4 = fi & 255;
                const float4 v = *reinterpret_cast<const float4*>(
                    xf + (size_t)toks[row] * HIDDEN + kb * 1024 + col4 * 4);
                const int c = col4 * 4;
                xs[row][c + 0] = (__bf16)v.x; xs[row][c + 1] = (__bf16)v.y;
                xs[row][c + 2] = (__bf16)v.z; xs[row][c + 3] = (__bf16)v.w;
            }
        }
        __syncthreads();

        for (int kk = 0; kk < 1024; kk += 32) {
            const v16bf a = *reinterpret_cast<const v16bf*>(&xs[ln16][kk + khalf]);
            const int ks  = (kb * 1024 + kk) >> 5;
#pragma unroll
            for (int nn = 0; nn < 6; ++nn) {
                const int ntg = wave * 6 + nn;
                v16bf bg, bu;
                if constexpr (PRE) {
                    const size_t fo = (((size_t)(e * (INTER >> 4) + ntg)) * KSGU + ks) * 32 + lane;
                    bg = *reinterpret_cast<const v16bf*>((const __bf16*)w_gate + fo * 16);
                    bu = *reinterpret_cast<const v16bf*>((const __bf16*)w_up   + fo * 16);
                } else {
                    const int n = ntg * 16 + ln16;
                    const size_t off = (size_t)e * INTER * HIDDEN + (size_t)n * HIDDEN
                                     + kb * 1024 + kk + khalf;
                    bg = cvt16((const float*)w_gate + off);
                    bu = cvt16((const float*)w_up   + off);
                }
                cg[nn] = wmma_bf16(a, bg, cg[nn]);
                cu[nn] = wmma_bf16(a, bu, cu[nn]);
            }
        }
        __syncthreads();
    }

    // ---- silu(g)*u, fold routing weight ----------------------------------
#pragma unroll
    for (int nn = 0; nn < 6; ++nn) {
        const int n = (wave * 6 + nn) * 16 + ln16;
#pragma unroll
        for (int r = 0; r < 8; ++r) {
            const int m = r + mhalf;
            const float g = cg[nn][r];
            const float u = cu[nn][r];
            const float s = g / (1.0f + __expf(-g));
            act[m][n] = (__bf16)(s * u * twt[m]);
        }
    }
    __syncthreads();

    // ---- down projection + atomic scatter --------------------------------
    for (int nt2 = 0; nt2 < 16; ++nt2) {
        const int ntg = wave * 16 + nt2;                 // 0..127 N-tiles of H
        v8f cy = (v8f)0.0f;
        for (int ks = 0; ks < KSD; ++ks) {
            const v16bf a = *reinterpret_cast<const v16bf*>(&act[ln16][ks * 32 + khalf]);
            v16bf b;
            if constexpr (PRE) {
                const size_t fo = (((size_t)(e * (HIDDEN >> 4) + ntg)) * KSD + ks) * 32 + lane;
                b = *reinterpret_cast<const v16bf*>((const __bf16*)w_down + fo * 16);
            } else {
                const size_t off = (size_t)e * HIDDEN * INTER
                                 + (size_t)(ntg * 16 + ln16) * INTER + ks * 32 + khalf;
                b = cvt16((const float*)w_down + off);
            }
            cy = wmma_bf16(a, b, cy);
        }
        const int n = ntg * 16 + ln16;
#pragma unroll
        for (int r = 0; r < 8; ++r) {
            const int m = r + mhalf;
            __hip_atomic_fetch_add(&out[(size_t)toks[m] * HIDDEN + n], cy[r],
                                   __ATOMIC_RELAXED, __HIP_MEMORY_SCOPE_AGENT);
        }
    }
}

// ---------------------------------------------------------------------------
extern "C" void kernel_launch(void* const* d_in, const int* in_sizes, int n_in,
                              void* d_out, int out_size, void* d_ws, size_t ws_size,
                              hipStream_t stream) {
    const float* x   = (const float*)d_in[0];
    const float* gw  = (const float*)d_in[1];
    const float* wg  = (const float*)d_in[2];
    const float* wu  = (const float*)d_in[3];
    const float* wd  = (const float*)d_in[4];
    float*       out = (float*)d_out;

    char* ws = (char*)d_ws;
    size_t off = 0;
    auto alloc = [&](size_t bytes) -> char* {
        char* p = ws + off;
        off = (off + bytes + 255) & ~(size_t)255;
        return p;
    };
    int*   cnt  = (int*)  alloc(NEXP * sizeof(int));
    int*   btok = (int*)  alloc((size_t)NEXP * NTOK * sizeof(int));
    float* bw   = (float*)alloc((size_t)NEXP * NTOK * sizeof(float));
    const size_t baseNeed = off;

    const size_t nW   = (size_t)NEXP * INTER * HIDDEN;   // per gate/up matrix
    const size_t nX   = (size_t)NTOK * HIDDEN;
    __bf16* xbf  = (__bf16*)alloc(nX * sizeof(__bf16));
    __bf16* wgb  = (__bf16*)alloc(nW * sizeof(__bf16));
    __bf16* wub  = (__bf16*)alloc(nW * sizeof(__bf16));
    __bf16* wdb  = (__bf16*)alloc(nW * sizeof(__bf16));
    const bool pre = (ws_size >= off) && (ws_size >= baseNeed);

    zero_kernel<<<2048, 256, 0, stream>>>(out, nX, cnt);
    route_kernel<<<NTOK / 8, 256, 0, stream>>>(x, gw, cnt, btok, bw);

    dim3 grid(NTOK / TILE_M, NEXP);
    if (pre) {
        convert_x_kernel<<<2048, 256, 0, stream>>>(x, xbf, nX / 4);
        swizzle_w_kernel<<<4096, 256, 0, stream>>>(wg, wgb, INTER,  HIDDEN, nW / 16);
        swizzle_w_kernel<<<4096, 256, 0, stream>>>(wu, wub, INTER,  HIDDEN, nW / 16);
        swizzle_w_kernel<<<4096, 256, 0, stream>>>(wd, wdb, HIDDEN, INTER,  nW / 16);
        expert_ffn_kernel<true><<<grid, 256, 0, stream>>>(
            xbf, wgb, wub, wdb, cnt, btok, bw, out);
    } else {
        expert_ffn_kernel<false><<<grid, 256, 0, stream>>>(
            x, wg, wu, wd, cnt, btok, bw, out);
    }
}
